// inter_prediction_40312563040278
// MI455X (gfx1250) — matrix operand order repaired and checked by reading, not compiled, not run on hardware
//
#include <hip/hip_runtime.h>

typedef __attribute__((ext_vector_type(16))) _Float16 v16h;
typedef __attribute__((ext_vector_type(8)))  float    v8f;

#define WAVES 16
#define BLK   (WAVES * 32)

// CDNA5: within-wave LDS RAW fence (split dep counters: DScnt)
__device__ __forceinline__ void wave_lds_fence() {
  __builtin_amdgcn_wave_barrier();
  asm volatile("s_wait_dscnt 0x0" ::: "memory");
  __builtin_amdgcn_wave_barrier();
}

// A fragment (16x32 f16, MxK) from row-major LDS tile; per ISA 7.12.2:
// lanes 0-15: m=lane, VGPR0-3 = K 0..7, VGPR4-7 = K 16..23
// lanes16-31: m=lane-16, VGPR0-3 = K 8..15, VGPR4-7 = K 24..31
// -> two contiguous 8-half (16B) runs per lane.
template <int STRIDE>
__device__ __forceinline__ v16h load_A(const _Float16* sm, int kOff) {
  const int lane = threadIdx.x & 31;
  const int m = lane & 15, hh = lane >> 4;
  const _Float16* p0 = sm + m * STRIDE + kOff + hh * 8;
  v16h a;
#pragma unroll
  for (int e = 0; e < 8; ++e) a[e] = p0[e];
#pragma unroll
  for (int e = 0; e < 8; ++e) a[8 + e] = p0[16 + e];
  return a;
}

// B fragment from pre-swizzled LDS (fragment-major: [lane][16 halfs] contiguous)
__device__ __forceinline__ v16h load_B(const _Float16* frag) {
  const int lane = threadIdx.x & 31;
  const _Float16* p = frag + lane * 16;
  v16h b;
#pragma unroll
  for (int e = 0; e < 16; ++e) b[e] = p[e];
  return b;
}

__device__ __forceinline__ v8f wmma_f16(v16h a, v16h b, v8f c) {
  return __builtin_amdgcn_wmma_f32_16x16x32_f16(false, a, false, b, (short)0, c,
                                                false, false);
}

// Convert fp32 weight W[Ksrc][Nsrc] (row-major) into f16 B-fragments in LDS,
// zero-padding K to KC*32 and N to NT*16. Fragment f=kc*NT+nt; within a
// fragment, lane n&15 is the column, (e>>3)*16+(lane>>4)*8+(e&7) is the K offset, so
// element order is exactly B-layout (two K halves per lane, pairs packed).
__device__ __forceinline__ void pack_B(_Float16* dst, const float* src,
                                       int Ksrc, int Nsrc, int KC, int NT,
                                       int tid) {
  const int tot = KC * NT * 512;
  for (int i = tid; i < tot; i += BLK) {
    int e  = i & 15;
    int ln = (i >> 4) & 31;
    int f  = i >> 9;
    int kc = f / NT, nt = f % NT;
    int k = kc * 32 + ((e >> 3) << 4) + ((ln >> 4) << 3) + (e & 7);
    int n = nt * 16 + (ln & 15);
    float v = (k < Ksrc && n < Nsrc) ? src[k * Nsrc + n] : 0.0f;
    dst[i] = (_Float16)v;
  }
}

__global__ __launch_bounds__(BLK, 1) void fused_inter_prediction(
    const float* __restrict__ xyz, const float* __restrict__ xyz_nn,
    const float* __restrict__ points, const int* __restrict__ knn,
    const float* __restrict__ dists, const unsigned char* __restrict__ mask,
    const float* __restrict__ w_dk1, const float* __restrict__ b_dk1,
    const float* __restrict__ w_dk2, const float* __restrict__ b_dk2,
    const float* __restrict__ w_m1, const float* __restrict__ b_m1,
    const float* __restrict__ w_m2, const float* __restrict__ b_m2,
    const float* __restrict__ w_sc, const float* __restrict__ b_sc,
    float* __restrict__ out_xyz, float* __restrict__ out_feat, int nPts) {
  // f16 weights in B-fragment order
  __shared__ _Float16 w1f[3 * 2 * 512];   // [96x32]
  __shared__ _Float16 w2f[1 * 5 * 512];   // [32x80]
  __shared__ _Float16 wm1f[3 * 4 * 512];  // [96x64]
  __shared__ _Float16 wm2f[2 * 4 * 512];  // [64x64]
  __shared__ _Float16 wscf[2 * 4 * 512];  // [64x64]
  // per-wave tiles
  __shared__ _Float16 gp_sm[WAVES][16 * 96];  // grouped feats, K-padded
  __shared__ _Float16 h_sm[WAVES][16 * 32];   // hidden of edge MLP
  // per-block tiles (16 points = M rows)
  __shared__ _Float16 agg_sm[16 * 96];
  __shared__ _Float16 pts_sm[16 * 64];
  __shared__ _Float16 t_sm[16 * 64];
  __shared__ float sc_sm[16 * 64];

  const int tid = threadIdx.x;
  const int wave = tid >> 5;
  const int lane = tid & 31;
  const int col16 = lane & 15;
  const int hh = lane >> 4;

  // pass-through xyz output (48 elems/block * 6250 blocks = 300000)
  {
    int xi = blockIdx.x * 48 + tid;
    if (tid < 48 && xi < nPts * 3) out_xyz[xi] = xyz[xi];
  }

  pack_B(w1f, w_dk1, 68, 32, 3, 2, tid);
  pack_B(w2f, w_dk2, 32, 68, 1, 5, tid);
  pack_B(wm1f, w_m1, 68, 64, 3, 4, tid);
  pack_B(wm2f, w_m2, 64, 64, 2, 4, tid);
  pack_B(wscf, w_sc, 64, 64, 2, 4, tid);
  for (int i = tid; i < 16 * 96; i += BLK) agg_sm[i] = (_Float16)0.0f;

  // ---- stage 1 gather: wave handles point p, lane pair (L, L^16) row k ----
  const int p = blockIdx.x * WAVES + wave;
  const bool pv = p < nPts;
  const int pc = pv ? p : 0;
  const int k = col16;
  const int nb = knn[pc * 16 + k];

  _Float16* gp = gp_sm[wave];
  {  // zero the K-padding cols 68..95 (14 cols per lane of the pair)
    int c0 = 68 + hh * 14;
#pragma unroll
    for (int c = 0; c < 14; ++c) gp[k * 96 + c0 + c] = (_Float16)0.0f;
  }
  if (hh == 0) {
    gp[k * 96 + 0] = (_Float16)(xyz_nn[nb * 3 + 0] - xyz[pc * 3 + 0]);
    gp[k * 96 + 1] = (_Float16)(xyz_nn[nb * 3 + 1] - xyz[pc * 3 + 1]);
    gp[k * 96 + 2] = (_Float16)(xyz_nn[nb * 3 + 2] - xyz[pc * 3 + 2]);
    gp[k * 96 + 3] = (_Float16)dists[pc * 16 + k];
  }
  {  // 32 gathered channels per lane, 128B vector loads (L2-resident)
    const float4* prow = (const float4*)(points + (size_t)nb * 64 + hh * 32);
#pragma unroll
    for (int j = 0; j < 8; ++j) {
      float4 f = prow[j];
      int c = 4 + hh * 32 + j * 4;
      gp[k * 96 + c + 0] = (_Float16)f.x;
      gp[k * 96 + c + 1] = (_Float16)f.y;
      gp[k * 96 + c + 2] = (_Float16)f.z;
      gp[k * 96 + c + 3] = (_Float16)f.w;
    }
  }
  const unsigned mbits = (unsigned)__ballot(mask[pc * 16 + k] != 0) & 0xFFFFu;

  {  // own-point features for the shortcut matmul (row = wave)
    float2 f2 = ((const float2*)(points + (size_t)pc * 64))[lane];
    pts_sm[wave * 64 + lane * 2 + 0] = (_Float16)f2.x;
    pts_sm[wave * 64 + lane * 2 + 1] = (_Float16)f2.y;
  }

  __syncthreads();  // weights + gp visible

  // ---- matmul1: h = relu(gp[16x96] @ W1[96x32] + b1) ----
#pragma unroll
  for (int nt = 0; nt < 2; ++nt) {
    float bb = b_dk1[nt * 16 + col16];
    v8f acc;
#pragma unroll
    for (int v = 0; v < 8; ++v) acc[v] = bb;
#pragma unroll
    for (int kc = 0; kc < 3; ++kc) {
      v16h a = load_A<96>(gp, kc * 32);
      v16h b = load_B(w1f + (kc * 2 + nt) * 512);
      acc = wmma_f16(a, b, acc);
    }
#pragma unroll
    for (int v = 0; v < 8; ++v)
      h_sm[wave][(hh * 8 + v) * 32 + nt * 16 + col16] =
          (_Float16)fmaxf(acc[v], 0.0f);
  }
  wave_lds_fence();

  // ---- matmul2 + masked softmax over neighbors + aggregation ----
  const v16h ah = load_A<32>(h_sm[wave], 0);
#pragma unroll
  for (int nt = 0; nt < 5; ++nt) {
    const int c = nt * 16 + col16;
    float bb = (c < 68) ? b_dk2[c] : 0.0f;
    v8f d;
#pragma unroll
    for (int v = 0; v < 8; ++v) d[v] = bb;
    d = wmma_f16(ah, load_B(w2f + nt * 512), d);

    // column (= channel) reduction over 16 neighbor rows: 8 in-lane + lane^16
    float mx = -3.402823466e+38f;
#pragma unroll
    for (int v = 0; v < 8; ++v) mx = fmaxf(mx, d[v]);
    mx = fmaxf(mx, __shfl_xor(mx, 16, 32));
    float e8[8], s = 0.0f;
#pragma unroll
    for (int v = 0; v < 8; ++v) {
      int r = hh * 8 + v;
      float ev = __expf(d[v] - mx) * (((mbits >> r) & 1u) ? 1.0f : 0.0f);
      e8[v] = ev;
      s += ev;
    }
    s += __shfl_xor(s, 16, 32);
    const float inv = 1.0f / (s + 1e-8f);
    float a = 0.0f;
#pragma unroll
    for (int v = 0; v < 8; ++v) {
      int r = hh * 8 + v;
      a += e8[v] * inv * (float)gp[r * 96 + c];
    }
    a += __shfl_xor(a, 16, 32);
    if (hh == 0 && c < 68 && pv) agg_sm[wave * 96 + c] = (_Float16)a;
  }
  __syncthreads();

  // ---- stage 2: 16 points batched as M=16; waves 0-3 main MLP, 4-7 shortcut
  if (wave < 4) {
    const int nt = wave;
    float bb = b_m1[nt * 16 + col16];
    v8f acc;
#pragma unroll
    for (int v = 0; v < 8; ++v) acc[v] = bb;
#pragma unroll
    for (int kc = 0; kc < 3; ++kc) {
      v16h a = load_A<96>(agg_sm, kc * 32);
      acc = wmma_f16(a, load_B(wm1f + (kc * 4 + nt) * 512), acc);
    }
#pragma unroll
    for (int v = 0; v < 8; ++v)
      t_sm[(hh * 8 + v) * 64 + nt * 16 + col16] = (_Float16)fmaxf(acc[v], 0.0f);
  } else if (wave < 8) {
    const int nt = wave - 4;
    float bb = b_sc[nt * 16 + col16] + b_m2[nt * 16 + col16];
    v8f acc;
#pragma unroll
    for (int v = 0; v < 8; ++v) acc[v] = bb;
#pragma unroll
    for (int kc = 0; kc < 2; ++kc) {
      v16h a = load_A<64>(pts_sm, kc * 32);
      acc = wmma_f16(a, load_B(wscf + (kc * 4 + nt) * 512), acc);
    }
#pragma unroll
    for (int v = 0; v < 8; ++v)
      sc_sm[(hh * 8 + v) * 64 + nt * 16 + col16] = acc[v];
  }
  __syncthreads();

  if (wave < 4) {
    const int nt = wave;
    v8f acc;
#pragma unroll
    for (int v = 0; v < 8; ++v)
      acc[v] = sc_sm[(hh * 8 + v) * 64 + nt * 16 + col16];
#pragma unroll
    for (int kc = 0; kc < 2; ++kc) {
      v16h a = load_A<64>(t_sm, kc * 32);
      acc = wmma_f16(a, load_B(wm2f + (kc * 4 + nt) * 512), acc);
    }
    const int c = nt * 16 + col16;
#pragma unroll
    for (int v = 0; v < 8; ++v) {
      int pp = blockIdx.x * 16 + hh * 8 + v;
      if (pp < nPts) out_feat[(size_t)pp * 64 + c] = acc[v];
    }
  }
}

extern "C" void kernel_launch(void* const* d_in, const int* in_sizes, int n_in,
                              void* d_out, int out_size, void* d_ws,
                              size_t ws_size, hipStream_t stream) {
  const float* xyz = (const float*)d_in[0];
  const float* xyz_nn = (const float*)d_in[1];
  const float* points = (const float*)d_in[2];
  const int* knn = (const int*)d_in[3];
  const float* dists = (const float*)d_in[4];
  const unsigned char* mask = (const unsigned char*)d_in[5];  // jnp bool_ = u8
  const float* w_dk1 = (const float*)d_in[6];
  const float* b_dk1 = (const float*)d_in[7];
  const float* w_dk2 = (const float*)d_in[8];
  const float* b_dk2 = (const float*)d_in[9];
  const float* w_m1 = (const float*)d_in[10];
  const float* b_m1 = (const float*)d_in[11];
  const float* w_m2 = (const float*)d_in[12];
  const float* b_m2 = (const float*)d_in[13];
  const float* w_sc = (const float*)d_in[14];
  const float* b_sc = (const float*)d_in[15];

  const int nPts = in_sizes[0] / 3;  // [B=1, N, 3]
  float* out_xyz = (float*)d_out;
  float* out_feat = out_xyz + in_sizes[0];

  const int blocks = (nPts + WAVES - 1) / WAVES;  // 6250 for N=100000
  hipLaunchKernelGGL(fused_inter_prediction, dim3(blocks), dim3(BLK), 0, stream,
                     xyz, xyz_nn, points, knn, dists, mask, w_dk1, b_dk1, w_dk2,
                     b_dk2, w_m1, b_m1, w_m2, b_m2, w_sc, b_sc, out_xyz,
                     out_feat, nPts);
}
